// ViT_8083128451527
// MI455X (gfx1250) — matrix operand order repaired and checked by reading, not compiled, not run on hardware
//
#include <hip/hip_runtime.h>
#include <hip/hip_bf16.h>

// ---------------------------------------------------------------------------
// Types for CDNA5 WMMA (wave32): D(f32 16x16) = A(bf16 16x32) * B(bf16 32x16) + C
// ---------------------------------------------------------------------------
typedef __bf16 bf16x16 __attribute__((ext_vector_type(16)));
typedef float  floatx8 __attribute__((ext_vector_type(8)));
typedef float  floatx4 __attribute__((ext_vector_type(4)));
typedef unsigned uintx4 __attribute__((ext_vector_type(4)));

__device__ __forceinline__ floatx8 wmma_bf16(bf16x16 a, bf16x16 b, floatx8 c) {
    // 8 args: (neg_a, A, neg_b, B, c_mod, C, reuse_a, reuse_b)
    return __builtin_amdgcn_wmma_f32_16x16x32_bf16(false, a, false, b, (short)0, c, false, false);
}

__device__ __forceinline__ unsigned short f2bf(float f) {
    union { float f; unsigned u; } x; x.f = f;
    unsigned r = x.u + 0x7FFFu + ((x.u >> 16) & 1u);   // round-to-nearest-even
    return (unsigned short)(r >> 16);
}
__device__ __forceinline__ float bf2f(unsigned short h) {
    union { unsigned u; float f; } x; x.u = ((unsigned)h) << 16; return x.f;
}

// A-matrix 16x32 bf16 fragment from row-major bf16 memory (row m, ld elems/row).
// ISA layout: lane L -> row m=L&15; per-lane K offsets are two contiguous runs of
// 8 elements starting at hi8 and 16+hi8 (hi8 = 8*(L>=16))  ->  2 x b128 loads.
__device__ __forceinline__ bf16x16 load_a_frag(const unsigned short* base, int ld, int m, int k0, int lane) {
    int hi8 = (lane >> 4) * 8;
    const unsigned short* row = base + (size_t)m * ld + k0 + hi8;
    union { bf16x16 v; uintx4 q[2]; } r;
    r.q[0] = *(const uintx4*)(row);
    r.q[1] = *(const uintx4*)(row + 16);
    return r.v;
}

// Same A layout, gathering fp32 rows: branch-free (safe pointer + 0/1 mask),
// 4 x global_load_b128 + hardware f32->bf16 converts.
__device__ __forceinline__ bf16x16 load_a_frag_f32(const float* rp, float msk, int k0, int lane) {
    int hi8 = (lane >> 4) * 8;
    const float* p = rp + k0 + hi8;
    floatx4 a0 = *(const floatx4*)(p);
    floatx4 a1 = *(const floatx4*)(p + 4);
    floatx4 a2 = *(const floatx4*)(p + 16);
    floatx4 a3 = *(const floatx4*)(p + 20);
    union { bf16x16 v; __bf16 h[16]; } r;
#pragma unroll
    for (int i = 0; i < 4; ++i) {
        r.h[i]      = (__bf16)(a0[i] * msk);
        r.h[4 + i]  = (__bf16)(a1[i] * msk);
        r.h[8 + i]  = (__bf16)(a2[i] * msk);
        r.h[12 + i] = (__bf16)(a3[i] * msk);
    }
    return r.v;
}

// B-matrix 32x16 bf16 fragment where memory holds B transposed: Bsrc[n][k].
// lane L -> col n=L&15; K runs of 8 at k0+hi8 and k0+16+hi8 (hi8 = 8*(L>=16)) -> 2 x b128.
__device__ __forceinline__ bf16x16 load_bT_frag(const unsigned short* base, int ld, int nbase, int k0, int lane) {
    const unsigned short* p = base + (size_t)(nbase + (lane & 15)) * ld + k0 + ((lane >> 4) << 3);
    union { bf16x16 v; uintx4 q[2]; } r;
    r.q[0] = *(const uintx4*)(p);
    r.q[1] = *(const uintx4*)(p + 16);
    return r.v;
}

// B fragment with only K=0..15 valid (elements 0..7 = K hi8..hi8+7, elements 8..15 zero) — used for K=16 PV steps.
__device__ __forceinline__ bf16x16 load_bT_frag_k16(const unsigned short* base, int ld, int nbase, int k0, int lane) {
    union { bf16x16 v; uintx4 q[2]; } r;
    {
        const unsigned short* p = base + (size_t)(nbase + (lane & 15)) * ld + k0 + ((lane >> 4) << 3);
        r.q[0] = *(const uintx4*)(p);
        r.q[1] = (uintx4){0u, 0u, 0u, 0u};
    }
    return r.v;
}

// P(16x16 f32 in LDS, row-major stride 16) -> A fragment with kmax=16 (upper half 0).
__device__ __forceinline__ bf16x16 load_p_frag(const float* pscr, int lane) {
    union { bf16x16 v; __bf16 h[16]; } r;
    int m = lane & 15;
    int hi8 = (lane >> 4) * 8;
    const float* row = pscr + m * 16 + hi8;   // 8 valid consecutive floats
#pragma unroll
    for (int i = 0; i < 8; ++i) {
        r.h[i]     = (__bf16)row[i];
        r.h[8 + i] = (__bf16)0.0f;
    }
    return r.v;
}

__device__ __forceinline__ floatx8 zero8() {
    floatx8 z = { 0.f, 0.f, 0.f, 0.f, 0.f, 0.f, 0.f, 0.f };
    return z;
}

// ---------------------------------------------------------------------------
// Problem constants: B=8, H=W=64, C=768, WS=14 -> 5x5 windows, 200 windows,
// N=196 tokens (padded to 208 = 13*16), 12 heads, hd=64.
// ---------------------------------------------------------------------------
#define NWIN   200
#define NPAD   208
#define NTOK   196
#define NHEAD  12
#define HD     64
#define CDIM   768
#define C3     2304

// ============================ kernel 1: weights -> bf16 =====================
__global__ void cvt_weights(const float* __restrict__ qkvw, const float* __restrict__ projw,
                            unsigned short* __restrict__ wq, unsigned short* __restrict__ wp) {
    int i = blockIdx.x * blockDim.x + threadIdx.x;
    if (i < C3 * CDIM)   wq[i] = f2bf(qkvw[i]);
    if (i < CDIM * CDIM) wp[i] = f2bf(projw[i]);
}

// ====== kernel 2: window-partition + QKV GEMM + bias + axial RoPE epilogue ==
// grid.x = NWIN * 36 (col blocks of 64 over 2304), block = 256 (8 waves).
// Each wave owns one 16-row M-tile and accumulates all 4 N-tiles of the block's
// 64 columns at once: per K-step 1 A-frag + 4 B-frags -> 4 WMMAs.
__global__ __launch_bounds__(256) void qkv_rope(
    const float* __restrict__ x, const unsigned short* __restrict__ wq,
    const float* __restrict__ qkvb,
    unsigned short* __restrict__ qbuf, unsigned short* __restrict__ kbuf,
    unsigned short* __restrict__ vtbuf) {
    int blk = blockIdx.x;
    int w  = blk / 36;
    int cb = blk % 36;
    int b  = w / 25, wy = (w % 25) / 5, wx = w % 5;
    int wave = threadIdx.x >> 5, lane = threadIdx.x & 31;
    int n  = lane & 15;
    int hi = lane >> 4;

    for (int mt = wave; mt < 13; mt += 8) {
        // branch-free gather setup: safe pointer + 0/1 mask
        int arow = mt * 16 + n;
        const float* rp = x;
        float msk = 0.f;
        if (arow < NTOK) {
            int ty = arow / 14, tx = arow % 14;
            int y = wy * 14 + ty, xx = wx * 14 + tx;
            if (y < 64 && xx < 64) {
                rp = x + (((size_t)b * 64 + y) * 64 + xx) * CDIM;
                msk = 1.f;
            }
        }
        floatx8 acc[4] = { zero8(), zero8(), zero8(), zero8() };
#pragma unroll 2
        for (int k0 = 0; k0 < CDIM; k0 += 32) {
            bf16x16 a = load_a_frag_f32(rp, msk, k0, lane);
#pragma unroll
            for (int nt = 0; nt < 4; ++nt)
                acc[nt] = wmma_bf16(a, load_bT_frag(wq, CDIM, cb * 64 + nt * 16, k0, lane), acc[nt]);
        }
        // epilogue: bias + RoPE(q,k) + scatter to head-major bf16 buffers
#pragma unroll
        for (int nt = 0; nt < 4; ++nt) {
            int o  = cb * 64 + nt * 16 + n;
            int sec = o / CDIM;                // 0=q 1=k 2=v
            int oc  = o % CDIM;
            int h   = oc / HD, d = oc % HD;
            float bias = qkvb[o];
            size_t whb = (size_t)(w * NHEAD + h);
#pragma unroll
            for (int r = 0; r < 8; ++r) {
                int tok = mt * 16 + r + 8 * hi;
                float val = (tok < NTOK) ? (acc[nt][r] + bias) : 0.f;
                if (sec < 2) {
                    // axial RoPE: pair p = d>>1; p<16 -> t_x*freq[p], else t_y*freq[p-16]
                    int p = d >> 1;
                    int fi = p & 15;
                    int tt = (tok < NTOK) ? tok : 0;
                    float pos = (p < 16) ? (float)(tt % 14) : (float)(tt / 14);
                    float fr  = __powf(10000.f, -(float)fi * (1.f / 16.f));
                    float s, c;
                    __sincosf(pos * fr, &s, &c);
                    float other = __shfl_xor(val, 1, 32);   // complex-pair partner
                    float res = (d & 1) ? (other * s + val * c)   // xi' = xr*sin + xi*cos
                                        : (val * c - other * s);  // xr' = xr*cos - xi*sin
                    val = (tok < NTOK) ? res : 0.f;
                }
                unsigned short bv = f2bf(val);
                if (sec == 0)      qbuf[(whb * NPAD + tok) * HD + d] = bv;
                else if (sec == 1) kbuf[(whb * NPAD + tok) * HD + d] = bv;
                else               vtbuf[(whb * HD + d) * NPAD + tok] = bv;   // v transposed
            }
        }
    }
}

// ================= kernel 3: flash attention per (window, head) =============
// grid.x = NWIN*NHEAD = 2400, block = 256 (8 waves)
__global__ __launch_bounds__(256) void attn_flash(
    const unsigned short* __restrict__ qbuf, const unsigned short* __restrict__ kbuf,
    const unsigned short* __restrict__ vtbuf,
    const float* __restrict__ relh, const float* __restrict__ relw,
    unsigned short* __restrict__ aout) {
    __shared__ float s_relh[NTOK * 14];
    __shared__ float s_relw[NTOK * 14];
    __shared__ float s_p[8][16 * 16];

    int wh = blockIdx.x;
    int w  = wh / NHEAD, hh = wh % NHEAD;
    const unsigned short* q  = qbuf  + (size_t)wh * NPAD * HD;
    const unsigned short* k  = kbuf  + (size_t)wh * NPAD * HD;
    const unsigned short* vt = vtbuf + (size_t)wh * HD * NPAD;

    // decomposed rel-pos bias tables: rel_h[t][a] = sum_d q[t][d]*rel_pos_h[ty-a+13][d]
    for (int idx = threadIdx.x; idx < NTOK * 14 * 2; idx += 256) {
        int which = (idx >= NTOK * 14);
        int e = idx - which * NTOK * 14;
        int t = e / 14, a = e % 14;
        int rel = ((which ? (t % 14) : (t / 14)) - a) + 13;
        const float* rp = (which ? relw : relh) + rel * HD;
        const unsigned short* qr = q + t * HD;
        float s = 0.f;
#pragma unroll 8
        for (int d = 0; d < HD; ++d) s += bf2f(qr[d]) * rp[d];
        (which ? s_relw : s_relh)[e] = s;
    }
    __syncthreads();

    int wave = threadIdx.x >> 5, lane = threadIdx.x & 31;
    float* pscr = &s_p[wave][0];
    int n  = lane & 15;
    int hi = lane >> 4;

    for (int qt = wave; qt < 13; qt += 8) {                 // per-wave 16-token q tile
        floatx8 o0 = zero8(), o1 = zero8(), o2 = zero8(), o3 = zero8();
        float rmax[8], rsum[8];
#pragma unroll
        for (int r = 0; r < 8; ++r) { rmax[r] = -3.0e38f; rsum[r] = 0.f; }

        for (int kt = 0; kt < 13; ++kt) {
            // S = q_tile @ k_tile^T  (16x16, K=64 in two WMMA steps)
            floatx8 s = zero8();
            s = wmma_bf16(load_a_frag(q, HD, qt * 16 + n, 0,  lane),
                          load_bT_frag(k, HD, kt * 16, 0,  lane), s);
            s = wmma_bf16(load_a_frag(q, HD, qt * 16 + n, 32, lane),
                          load_bT_frag(k, HD, kt * 16, 32, lane), s);

            int j = kt * 16 + n;                            // key token of this lane
            int ja = (j < NTOK) ? (j / 14) : 0;
            int jb = (j < NTOK) ? (j % 14) : 0;
            float pl[8];
#pragma unroll
            for (int r = 0; r < 8; ++r) {
                int m = qt * 16 + r + 8 * hi;
                int tq = (m < NTOK) ? m : NTOK - 1;         // clamp pad rows (discarded later)
                pl[r] = (j < NTOK)
                      ? (s[r] * 0.125f + s_relh[tq * 14 + ja] + s_relw[tq * 14 + jb])
                      : -1e30f;                              // mask padded keys
            }
            // online softmax: row stats across the 16 lanes of each half-wave
#pragma unroll
            for (int r = 0; r < 8; ++r) {
                float v = pl[r];
                v = fmaxf(v, __shfl_xor(v, 1, 32));
                v = fmaxf(v, __shfl_xor(v, 2, 32));
                v = fmaxf(v, __shfl_xor(v, 4, 32));
                v = fmaxf(v, __shfl_xor(v, 8, 32));
                float newmax = fmaxf(rmax[r], v);
                float scale  = __expf(rmax[r] - newmax);
                rmax[r] = newmax;
                float p = (pl[r] > -1e29f) ? __expf(pl[r] - newmax) : 0.f;
                pl[r] = p;
                float ps = p;
                ps += __shfl_xor(ps, 1, 32);
                ps += __shfl_xor(ps, 2, 32);
                ps += __shfl_xor(ps, 4, 32);
                ps += __shfl_xor(ps, 8, 32);
                rsum[r] = rsum[r] * scale + ps;
                o0[r] *= scale; o1[r] *= scale; o2[r] *= scale; o3[r] *= scale;
            }
            // D-layout -> A-layout transpose of P through per-wave LDS tile
#pragma unroll
            for (int r = 0; r < 8; ++r) pscr[(r + 8 * hi) * 16 + n] = pl[r];
            asm volatile("s_wait_dscnt 0" ::: "memory");    // wave-internal LDS RAW
            bf16x16 pa = load_p_frag(pscr, lane);
            // O += P @ V  (K=16 step: upper K half zeroed)
            o0 = wmma_bf16(pa, load_bT_frag_k16(vt, NPAD, 0,  kt * 16, lane), o0);
            o1 = wmma_bf16(pa, load_bT_frag_k16(vt, NPAD, 16, kt * 16, lane), o1);
            o2 = wmma_bf16(pa, load_bT_frag_k16(vt, NPAD, 32, kt * 16, lane), o2);
            o3 = wmma_bf16(pa, load_bT_frag_k16(vt, NPAD, 48, kt * 16, lane), o3);
        }
        // normalize + store bf16 (layout: [window, token, h*64+d] for proj GEMM)
#pragma unroll
        for (int r = 0; r < 8; ++r) {
            int m = qt * 16 + r + 8 * hi;
            float inv = 1.f / rsum[r];
            float v0 = o0[r] * inv, v1 = o1[r] * inv, v2 = o2[r] * inv, v3 = o3[r] * inv;
            if (m >= NTOK) { v0 = v1 = v2 = v3 = 0.f; }
            size_t rowoff = ((size_t)w * NPAD + m) * CDIM + hh * HD;
            aout[rowoff + 0  + n] = f2bf(v0);
            aout[rowoff + 16 + n] = f2bf(v1);
            aout[rowoff + 32 + n] = f2bf(v2);
            aout[rowoff + 48 + n] = f2bf(v3);
        }
    }
}

// ========== kernel 4: out-proj GEMM + bias + window-unpartition/crop ========
// grid.x = NWIN * 12 (col blocks of 64 over 768), block = 256 (8 waves)
__global__ __launch_bounds__(256) void proj_scatter(
    const unsigned short* __restrict__ aout, const unsigned short* __restrict__ wp,
    const float* __restrict__ projb, float* __restrict__ out) {
    int blk = blockIdx.x;
    int w  = blk / 12;
    int cb = blk % 12;
    int b  = w / 25, wy = (w % 25) / 5, wx = w % 5;
    int wave = threadIdx.x >> 5, lane = threadIdx.x & 31;
    int n  = lane & 15;
    int hi = lane >> 4;
    const unsigned short* A = aout + (size_t)w * NPAD * CDIM;

    for (int mt = wave; mt < 13; mt += 8) {
        floatx8 acc[4] = { zero8(), zero8(), zero8(), zero8() };
#pragma unroll 2
        for (int k0 = 0; k0 < CDIM; k0 += 32) {
            bf16x16 a = load_a_frag(A, CDIM, mt * 16 + n, k0, lane);
#pragma unroll
            for (int nt = 0; nt < 4; ++nt)
                acc[nt] = wmma_bf16(a, load_bT_frag(wp, CDIM, cb * 64 + nt * 16, k0, lane), acc[nt]);
        }
#pragma unroll
        for (int nt = 0; nt < 4; ++nt) {
            int o = cb * 64 + nt * 16 + n;
            float bias = projb[o];
#pragma unroll
            for (int r = 0; r < 8; ++r) {
                int m = mt * 16 + r + 8 * hi;
                if (m < NTOK) {
                    int ty = m / 14, tx = m % 14;
                    int y = wy * 14 + ty, xx = wx * 14 + tx;
                    if (y < 64 && xx < 64)
                        out[(((size_t)b * 64 + y) * 64 + xx) * CDIM + o] = acc[nt][r] + bias;
                }
            }
        }
    }
}

// ---------------------------------------------------------------------------
extern "C" void kernel_launch(void* const* d_in, const int* in_sizes, int n_in,
                              void* d_out, int out_size, void* d_ws, size_t ws_size,
                              hipStream_t stream) {
    (void)in_sizes; (void)n_in; (void)out_size; (void)ws_size;
    const float* x     = (const float*)d_in[0];
    const float* qkvw  = (const float*)d_in[1];
    const float* qkvb  = (const float*)d_in[2];
    const float* projw = (const float*)d_in[3];
    const float* projb = (const float*)d_in[4];
    const float* relh  = (const float*)d_in[5];
    const float* relw  = (const float*)d_in[6];
    float* out = (float*)d_out;

    char* ws = (char*)d_ws;
    size_t off = 0;
    auto carve = [&](size_t bytes) {
        void* p = ws + off;
        off += (bytes + 255) & ~(size_t)255;
        return p;
    };
    unsigned short* wq    = (unsigned short*)carve((size_t)C3 * CDIM * 2);
    unsigned short* wp    = (unsigned short*)carve((size_t)CDIM * CDIM * 2);
    unsigned short* qbuf  = (unsigned short*)carve((size_t)NWIN * NHEAD * NPAD * HD * 2);
    unsigned short* kbuf  = (unsigned short*)carve((size_t)NWIN * NHEAD * NPAD * HD * 2);
    unsigned short* vtbuf = (unsigned short*)carve((size_t)NWIN * NHEAD * HD * NPAD * 2);
    unsigned short* abuf  = (unsigned short*)carve((size_t)NWIN * NPAD * CDIM * 2);

    cvt_weights<<<(C3 * CDIM + 255) / 256, 256, 0, stream>>>(qkvw, projw, wq, wp);
    qkv_rope<<<NWIN * 36, 256, 0, stream>>>(x, wq, qkvb, qbuf, kbuf, vtbuf);
    attn_flash<<<NWIN * NHEAD, 256, 0, stream>>>(qbuf, kbuf, vtbuf, relh, relw, abuf);
    proj_scatter<<<NWIN * 12, 256, 0, stream>>>(abuf, wp, projb, out);
}
